// HAFM_73916387164772
// MI455X (gfx1250) — compile-verified, matrix operand order checked by reading, not run on hardware
//
#include <hip/hip_runtime.h>
#include <math.h>

// ---------------------------------------------------------------------------
// Types for CDNA5 WMMA (wave32): v_wmma_f32_16x16x32_bf16
// ---------------------------------------------------------------------------
typedef __bf16 bf16_t;
typedef __attribute__((ext_vector_type(16))) __bf16 bf16x16;
typedef __attribute__((ext_vector_type(8)))  __bf16 bf16x8;
typedef __attribute__((ext_vector_type(8)))  float  v8f;

#define BQ 16
#define CC 256
#define HH 64
#define WW 64
#define NN 4096                 // H*W
#define MTOK (BQ * NN)          // 65536 token rows
#define CR 128
#define NHD 8
#define HIDN 1024
#define HALFN 512
#define NRR 16                  // reduced tokens per batch (4x4)

__device__ __forceinline__ float gelu_f(float x) {
    return 0.5f * x * (1.0f + erff(x * 0.70710678118654752f));
}

// ---------------------------------------------------------------------------
// Weight packing: Bm is [K,N] row-major f32 (or [N,K] if trans).
// Packed layout: Bp[((kt*N + n)*2 + half)*16 + i], i<8 -> K=kt*32+half*8+i,
// i>=8 -> K=kt*32+16+half*8+(i-8).  One contiguous 32B load per lane per frag.
// ---------------------------------------------------------------------------
__global__ void pack_b_k(const float* __restrict__ Wm, bf16_t* __restrict__ Bp,
                         int K, int N, int trans) {
    int idx = blockIdx.x * blockDim.x + threadIdx.x;
    if (idx >= K * N) return;
    int i    = idx & 15;
    int half = (idx >> 4) & 1;
    int rest = idx >> 5;
    int n    = rest % N;
    int kt   = rest / N;
    int k = kt * 32 + ((i < 8) ? (half * 8 + i) : (16 + half * 8 + (i - 8)));
    float v = trans ? Wm[(size_t)n * K + k] : Wm[(size_t)k * N + n];
    Bp[idx] = (bf16_t)v;
}

// ---------------------------------------------------------------------------
// WMMA GEMM: C[M,N] = act(A[M,K] @ B[K,N] + bias) + resid
// A bf16 row-major, Bp packed bf16. One wave -> 16x64 tile, 4 waves/block.
// ACT: 0 none, 1 gelu.  OUTMODE: 0 f32 row-major; 1 f32 scatter to NCHW
// (row = b*4096 + n, out[((b*N + col)<<12) + n]).
// ---------------------------------------------------------------------------
template <int ACT, int OUTMODE>
__global__ __launch_bounds__(128)
void gemm_wmma_k(const bf16_t* __restrict__ A, const bf16_t* __restrict__ Bp,
                 const float* __restrict__ bias, const float* __restrict__ resid,
                 float* __restrict__ outp, int M, int N, int K, int ldr) {
    int lane = threadIdx.x & 31;
    int wv   = threadIdx.x >> 5;
    int m0   = blockIdx.x * 64 + wv * 16;
    int n0   = blockIdx.y * 64;
    if (m0 >= M) return;
    int nl   = lane & 15;
    int half = lane >> 4;
    int mrow = m0 + nl;

    const bf16_t* arow = A + (size_t)mrow * K + half * 8;
    v8f acc[4] = {};

    for (int kt = 0; kt < K; kt += 32) {
        bf16x16 a;
        bf16x8 alo = *(const bf16x8*)(arow + kt);
        bf16x8 ahi = *(const bf16x8*)(arow + kt + 16);
#pragma unroll
        for (int i = 0; i < 8; ++i) { a[i] = alo[i]; a[i + 8] = ahi[i]; }

        const bf16_t* bbase = Bp + (size_t)(kt >> 5) * N * 32 + half * 16;
#pragma unroll
        for (int nt = 0; nt < 4; ++nt) {
            int n = n0 + nt * 16 + nl;
            const bf16_t* bp = bbase + (size_t)n * 32;
            bf16x16 b;
            bf16x8 blo = *(const bf16x8*)(bp);
            bf16x8 bhi = *(const bf16x8*)(bp + 8);
#pragma unroll
            for (int i = 0; i < 8; ++i) { b[i] = blo[i]; b[i + 8] = bhi[i]; }
            acc[nt] = __builtin_amdgcn_wmma_f32_16x16x32_bf16(
                false, a, false, b, (short)0, acc[nt], false, false);
        }
    }

#pragma unroll
    for (int nt = 0; nt < 4; ++nt) {
        int col = n0 + nt * 16 + nl;
#pragma unroll
        for (int r = 0; r < 8; ++r) {
            int row = m0 + half * 8 + r;
            float val = acc[nt][r];
            if (bias)  val += bias[col];
            if (ACT == 1) val = gelu_f(val);
            if (resid) val += resid[(size_t)row * ldr + col];
            if (OUTMODE == 0) {
                outp[(size_t)row * N + col] = val;
            } else {
                int bb = row >> 12, nn = row & 4095;
                outp[(((size_t)bb * N + col) << 12) + nn] = val;
            }
        }
    }
}

// ---------------------------------------------------------------------------
// LayerNorm (one wave per row).  GELU: apply gelu after affine.
// OUTMODE 0: bf16 row-major [M,D].  OUTMODE 1: f32 scatter to [B, D, 4096].
// ---------------------------------------------------------------------------
template <int GELU, int OUTMODE>
__global__ __launch_bounds__(128)
void layernorm_k(const float* __restrict__ in, int inStride, int inOff,
                 const float* __restrict__ w, const float* __restrict__ b,
                 float eps, int M, int D,
                 bf16_t* __restrict__ outB, float* __restrict__ outF) {
    int row  = blockIdx.x * 4 + (threadIdx.x >> 5);
    int lane = threadIdx.x & 31;
    if (row >= M) return;
    const float* x = in + (size_t)row * inStride + inOff;
    float s = 0.f, s2 = 0.f;
    for (int i = lane; i < D; i += 32) { float v = x[i]; s += v; s2 += v * v; }
#pragma unroll
    for (int off = 16; off; off >>= 1) {
        s  += __shfl_xor(s, off, 32);
        s2 += __shfl_xor(s2, off, 32);
    }
    float mean = s / D;
    float var  = s2 / D - mean * mean;
    float rstd = rsqrtf(var + eps);
    for (int i = lane; i < D; i += 32) {
        float v = (x[i] - mean) * rstd * w[i] + b[i];
        if (GELU) v = gelu_f(v);
        if (OUTMODE == 0) {
            outB[(size_t)row * D + i] = (bf16_t)v;
        } else {
            int bb = row >> 12, nn = row & 4095;
            outF[(((size_t)bb * D + i) << 12) + nn] = v;
        }
    }
}

// ---------------------------------------------------------------------------
// SDAM kernels
// ---------------------------------------------------------------------------
__global__ void pool_w_k(const float* __restrict__ x, float* __restrict__ out) {
    int idx = blockIdx.x * blockDim.x + threadIdx.x;        // B*C*H
    if (idx >= BQ * CC * HH) return;
    const float* p = x + (size_t)idx * WW;
    float s = 0.f;
#pragma unroll 8
    for (int i = 0; i < WW; ++i) s += p[i];
    out[idx] = s * (1.0f / WW);
}

__global__ void pool_h_k(const float* __restrict__ x, float* __restrict__ out) {
    int idx = blockIdx.x * blockDim.x + threadIdx.x;        // B*C*W
    if (idx >= BQ * CC * WW) return;
    int w  = idx & (WW - 1);
    int bc = idx >> 6;
    const float* p = x + (size_t)bc * NN + w;
    float s = 0.f;
#pragma unroll 8
    for (int i = 0; i < HH; ++i) s += p[i * WW];
    out[idx] = s * (1.0f / HH);
}

__global__ void dwconv1d_k7_k(const float* __restrict__ in,
                              const float* __restrict__ w,
                              float* __restrict__ out) {
    int idx = blockIdx.x * blockDim.x + threadIdx.x;        // B*C*64
    if (idx >= BQ * CC * 64) return;
    int l = idx & 63;
    int c = (idx >> 6) & (CC - 1);
    const float* p = in + (idx - l);
    float s = 0.f;
#pragma unroll
    for (int j = 0; j < 7; ++j) {
        int pos = l + j - 3;
        if (pos >= 0 && pos < 64) s += p[pos] * w[c * 7 + j];
    }
    out[idx] = s;
}

// GroupNorm(16 groups of 16 ch over L=64) + ReLU; one block per (b, group)
__global__ __launch_bounds__(256)
void gn_relu_k(const float* __restrict__ in, const float* __restrict__ w,
               const float* __restrict__ b, float* __restrict__ out) {
    int bg = blockIdx.x;
    int bb = bg >> 4, g = bg & 15;
    const float* p = in + ((size_t)bb * CC + g * 16) * 64;  // 1024 contiguous
    __shared__ float redS[8], redS2[8];
    float v[4], s = 0.f, s2 = 0.f;
#pragma unroll
    for (int i = 0; i < 4; ++i) {
        v[i] = p[threadIdx.x + i * 256];
        s += v[i]; s2 += v[i] * v[i];
    }
#pragma unroll
    for (int off = 16; off; off >>= 1) {
        s  += __shfl_xor(s, off, 32);
        s2 += __shfl_xor(s2, off, 32);
    }
    int wv = threadIdx.x >> 5, lane = threadIdx.x & 31;
    if (lane == 0) { redS[wv] = s; redS2[wv] = s2; }
    __syncthreads();
    float ts = 0.f, ts2 = 0.f;
#pragma unroll
    for (int i = 0; i < 8; ++i) { ts += redS[i]; ts2 += redS2[i]; }
    float mean = ts * (1.0f / 1024.0f);
    float var  = ts2 * (1.0f / 1024.0f) - mean * mean;
    float rstd = rsqrtf(var + 1e-5f);
#pragma unroll
    for (int i = 0; i < 4; ++i) {
        int el = threadIdx.x + i * 256;
        int c  = g * 16 + (el >> 6);
        float y = (v[i] - mean) * rstd * w[c] + b[c];
        out[((size_t)bb * CC + c) * 64 + (el & 63)] = fmaxf(y, 0.f);
    }
}

// x * xh[..h] * xw[..w], transposed to tokens t[b,n,c]
__global__ void gate_transpose_k(const float* __restrict__ x,
                                 const float* __restrict__ xh,
                                 const float* __restrict__ xw,
                                 float* __restrict__ t) {
    size_t idx = (size_t)blockIdx.x * blockDim.x + threadIdx.x;  // B*C*N
    if (idx >= (size_t)BQ * CC * NN) return;
    int    n  = (int)(idx & (NN - 1));
    size_t bc = idx >> 12;
    int    c  = (int)(bc & (CC - 1));
    int    b  = (int)(bc >> 8);
    int hh = n >> 6, ww = n & 63;
    float v = x[idx] * xh[bc * 64 + hh] * xw[bc * 64 + ww];
    t[((size_t)b * NN + n) * CC + c] = v;
}

// ---------------------------------------------------------------------------
// Spatial-reduction path (tiny)
// ---------------------------------------------------------------------------
// dw 4x4 stride 4 reading bf16 tokens as image -> red1 [B,C,16,16] f32
__global__ void red1_k(const bf16_t* __restrict__ tln, const float* __restrict__ w,
                       const float* __restrict__ b, float* __restrict__ out) {
    int idx = blockIdx.x * blockDim.x + threadIdx.x;    // B*C*16*16
    if (idx >= BQ * CC * 256) return;
    int xx = idx & 15, yy = (idx >> 4) & 15, c = (idx >> 8) & (CC - 1), bb = idx >> 16;
    float s = 0.f;
#pragma unroll
    for (int i = 0; i < 4; ++i)
#pragma unroll
        for (int j = 0; j < 4; ++j) {
            int h = yy * 4 + i, w4 = xx * 4 + j;
            float v = (float)tln[((size_t)bb * NN + h * WW + w4) * CC + c];
            s += v * w[c * 16 + i * 4 + j];
        }
    out[idx] = s + b[c];
}

// dw 4x4 stride 4 on [B,C,16,16] -> [B,C,4,4]
__global__ void red2_k(const float* __restrict__ in, const float* __restrict__ w,
                       const float* __restrict__ b, float* __restrict__ out) {
    int idx = blockIdx.x * blockDim.x + threadIdx.x;    // B*C*16
    if (idx >= BQ * CC * 16) return;
    int xx = idx & 3, yy = (idx >> 2) & 3, c = (idx >> 4) & (CC - 1), bb = idx >> 12;
    const float* p = in + ((size_t)bb * CC + c) * 256;
    float s = 0.f;
#pragma unroll
    for (int i = 0; i < 4; ++i)
#pragma unroll
        for (int j = 0; j < 4; ++j)
            s += p[(yy * 4 + i) * 16 + xx * 4 + j] * w[c * 16 + i * 4 + j];
    out[idx] = s + b[c];
}

// dw 3x3 pad1 on [B,C,4,4] -> transposed bf16 A-matrix [B*16, 256]
__global__ void dw3_t_k(const float* __restrict__ in, const float* __restrict__ w,
                        const float* __restrict__ b, bf16_t* __restrict__ out) {
    int idx = blockIdx.x * blockDim.x + threadIdx.x;    // B*C*16
    if (idx >= BQ * CC * 16) return;
    int xx = idx & 3, yy = (idx >> 2) & 3, c = (idx >> 4) & (CC - 1), bb = idx >> 12;
    const float* p = in + ((size_t)bb * CC + c) * 16;
    float s = 0.f;
#pragma unroll
    for (int i = -1; i <= 1; ++i)
#pragma unroll
        for (int j = -1; j <= 1; ++j) {
            int y = yy + i, x4 = xx + j;
            if ((unsigned)y < 4u && (unsigned)x4 < 4u)
                s += p[y * 4 + x4] * w[c * 9 + (i + 1) * 3 + (j + 1)];
        }
    out[((size_t)bb * NRR + yy * 4 + xx) * CC + c] = (bf16_t)(s + b[c]);
}

// CPE: v += dwconv3x3(v as [B,C,4,4])
__global__ void cpe_k(const float* __restrict__ vin, const float* __restrict__ w,
                      const float* __restrict__ b, float* __restrict__ vout) {
    int idx = blockIdx.x * blockDim.x + threadIdx.x;    // B*C*16
    if (idx >= BQ * CC * 16) return;
    int nr = idx & 15, c = (idx >> 4) & (CC - 1), bb = idx >> 12;
    int yy = nr >> 2, xx = nr & 3;
    float s = 0.f;
#pragma unroll
    for (int i = -1; i <= 1; ++i)
#pragma unroll
        for (int j = -1; j <= 1; ++j) {
            int y = yy + i, x4 = xx + j;
            if ((unsigned)y < 4u && (unsigned)x4 < 4u)
                s += vin[((size_t)bb * NRR + y * 4 + x4) * CC + c] *
                     w[c * 9 + (i + 1) * 3 + (j + 1)];
        }
    vout[((size_t)bb * NRR + nr) * CC + c] =
        vin[((size_t)bb * NRR + nr) * CC + c] + s + b[c];
}

// ---------------------------------------------------------------------------
// Fused attention over 16 reduced keys; K/V staged in LDS (24 KB / WG).
// One thread = one token; writes o bf16 [M, 256].
// ---------------------------------------------------------------------------
__global__ __launch_bounds__(256)
void attn_k(const float* __restrict__ q, const float* __restrict__ kbuf,
            const float* __restrict__ vbuf, bf16_t* __restrict__ o) {
    __shared__ float ks[NRR * CR];       // 8 KB
    __shared__ float vs[NRR * CC];       // 16 KB
    int b    = blockIdx.x >> 4;
    int nblk = blockIdx.x & 15;
    for (int i = threadIdx.x; i < NRR * CR; i += 256) ks[i] = kbuf[(size_t)b * NRR * CR + i];
    for (int i = threadIdx.x; i < NRR * CC; i += 256) vs[i] = vbuf[(size_t)b * NRR * CC + i];
    __syncthreads();
    int n = nblk * 256 + threadIdx.x;
    const float* qp = q + ((size_t)b * NN + n) * CR;
    bf16_t* op = o + ((size_t)b * NN + n) * CC;
    for (int h = 0; h < NHD; ++h) {
        float qv[16];
#pragma unroll
        for (int d = 0; d < 16; ++d) qv[d] = qp[h * 16 + d];
        float sc[16], mx = -1e30f;
#pragma unroll
        for (int m = 0; m < 16; ++m) {
            float s = 0.f;
#pragma unroll
            for (int d = 0; d < 16; ++d) s += qv[d] * ks[m * CR + h * 16 + d];
            s *= 0.25f;                  // SCALE
            sc[m] = s; mx = fmaxf(mx, s);
        }
        float sum = 0.f;
#pragma unroll
        for (int m = 0; m < 16; ++m) { sc[m] = __expf(sc[m] - mx); sum += sc[m]; }
        float inv = 1.0f / sum;
#pragma unroll
        for (int e = 0; e < 32; ++e) {
            float acc = 0.f;
#pragma unroll
            for (int m = 0; m < 16; ++m) acc += sc[m] * vs[m * CC + h * 32 + e];
            op[h * 32 + e] = (bf16_t)(acc * inv);
        }
    }
}

// ---------------------------------------------------------------------------
// Gated-MLP conv: prod[b,n,c] = x1[b,n,c] * (dw3x3(x2ln_img)[b,c,n] + gc_b[c])
// ---------------------------------------------------------------------------
__global__ void gconv_mul_k(const float* __restrict__ x2img,
                            const float* __restrict__ h,
                            const float* __restrict__ w,
                            const float* __restrict__ b,
                            bf16_t* __restrict__ prod) {
    size_t idx = (size_t)blockIdx.x * blockDim.x + threadIdx.x;  // B*512*4096
    if (idx >= (size_t)BQ * HALFN * NN) return;
    int    n  = (int)(idx & (NN - 1));
    size_t bc = idx >> 12;
    int    c  = (int)(bc & (HALFN - 1));
    int    bb = (int)(bc >> 9);
    int hh = n >> 6, ww = n & 63;
    const float* p = x2img + bc * NN;
    float s = 0.f;
#pragma unroll
    for (int i = -1; i <= 1; ++i)
#pragma unroll
        for (int j = -1; j <= 1; ++j) {
            int y = hh + i, x4 = ww + j;
            if ((unsigned)y < 64u && (unsigned)x4 < 64u)
                s += p[y * WW + x4] * w[c * 9 + (i + 1) * 3 + (j + 1)];
        }
    s += b[c];
    float x1 = h[((size_t)bb * NN + n) * HIDN + c];
    prod[((size_t)bb * NN + n) * HALFN + c] = (bf16_t)(x1 * s);
}

// ---------------------------------------------------------------------------
extern "C" void kernel_launch(void* const* d_in, const int* in_sizes, int n_in,
                              void* d_out, int out_size, void* d_ws, size_t ws_size,
                              hipStream_t stream) {
    (void)in_sizes; (void)n_in; (void)out_size; (void)ws_size;
    const float* x        = (const float*)d_in[0];
    const float* sda_w    = (const float*)d_in[1];
    const float* sda_gn_w = (const float*)d_in[2];
    const float* sda_gn_b = (const float*)d_in[3];
    const float* n1_w     = (const float*)d_in[4];
    const float* n1_b     = (const float*)d_in[5];
    const float* red_w    = (const float*)d_in[6];
    const float* red_b    = (const float*)d_in[7];
    const float* dw_w     = (const float*)d_in[8];
    const float* dw_b     = (const float*)d_in[9];
    const float* conv_w   = (const float*)d_in[10];
    const float* conv_b   = (const float*)d_in[11];
    const float* na_w     = (const float*)d_in[12];
    const float* na_b     = (const float*)d_in[13];
    const float* q_w      = (const float*)d_in[14];
    const float* k_w      = (const float*)d_in[15];
    const float* v_w      = (const float*)d_in[16];
    const float* cpe_w    = (const float*)d_in[17];
    const float* cpe_b    = (const float*)d_in[18];
    const float* proj_w   = (const float*)d_in[19];
    const float* proj_b   = (const float*)d_in[20];
    const float* n2_w     = (const float*)d_in[21];
    const float* n2_b     = (const float*)d_in[22];
    const float* fc1_w    = (const float*)d_in[23];
    const float* fc1_b    = (const float*)d_in[24];
    const float* gn_w     = (const float*)d_in[25];
    const float* gn_b     = (const float*)d_in[26];
    const float* gc_w     = (const float*)d_in[27];
    const float* gc_b     = (const float*)d_in[28];
    const float* fc2_w    = (const float*)d_in[29];
    const float* fc2_b    = (const float*)d_in[30];
    float* out = (float*)d_out;

    char* W = (char*)d_ws;
    const size_t MB = 1024ull * 1024ull;
    // Large buffers (explicit lifetimes; x2img aliases dead region [0,128MB))
    float*  t_res = (float*)(W + 0);          // 64MB  [M,256] f32 residual 1
    bf16_t* t_ln  = (bf16_t*)(W + 64 * MB);   // 32MB  [M,256] bf16
    float*  qbuf  = (float*)(W + 96 * MB);    // 32MB  [M,128] f32
    bf16_t* obuf  = (bf16_t*)(W + 128 * MB);  // 32MB  [M,256] bf16
    float*  t2    = (float*)(W + 160 * MB);   // 64MB  [M,256] f32 residual 2
    bf16_t* t2_ln = (bf16_t*)(W + 224 * MB);  // 32MB
    float*  hbuf  = (float*)(W + 256 * MB);   // 256MB [M,1024] f32
    bf16_t* prod  = (bf16_t*)(W + 512 * MB);  // 64MB  [M,512] bf16
    float*  x2img = (float*)(W + 0);          // 128MB alias (t/t_ln/q dead)
    char* S = W + 576 * MB;                   // small buffers
    float*  xm3   = (float*)(S + 0 * MB);
    float*  xm2   = (float*)(S + 1 * MB);
    float*  c1    = (float*)(S + 2 * MB);
    float*  c2    = (float*)(S + 3 * MB);
    float*  xhb   = (float*)(S + 4 * MB);
    float*  xwb   = (float*)(S + 5 * MB);
    float*  red1b = (float*)(S + 6 * MB);     // 4MB
    float*  red2b = (float*)(S + 10 * MB);
    bf16_t* dwt   = (bf16_t*)(S + 11 * MB);
    float*  cx    = (float*)(S + 12 * MB);
    bf16_t* xr    = (bf16_t*)(S + 13 * MB);
    float*  kbuf  = (float*)(S + 14 * MB);
    float*  vbuf  = (float*)(S + 15 * MB);
    float*  vbuf2 = (float*)(S + 16 * MB);
    bf16_t* qw_p  = (bf16_t*)(S + 17 * MB);
    bf16_t* kw_p  = (bf16_t*)(S + 18 * MB);
    bf16_t* vw_p  = (bf16_t*)(S + 19 * MB);
    bf16_t* cw_p  = (bf16_t*)(S + 20 * MB);
    bf16_t* pw_p  = (bf16_t*)(S + 21 * MB);
    bf16_t* f1w_p = (bf16_t*)(S + 22 * MB);   // 512KB
    bf16_t* f2w_p = (bf16_t*)(S + 23 * MB);

    // ---- pack weights to bf16 WMMA layout (once per launch) ----
    pack_b_k<<<(256 * 128 + 255) / 256, 256, 0, stream>>>(q_w,    qw_p,  256, 128, 0);
    pack_b_k<<<(128 * 128 + 255) / 256, 256, 0, stream>>>(k_w,    kw_p,  128, 128, 0);
    pack_b_k<<<(128 * 256 + 255) / 256, 256, 0, stream>>>(v_w,    vw_p,  128, 256, 0);
    pack_b_k<<<(256 * 128 + 255) / 256, 256, 0, stream>>>(conv_w, cw_p,  256, 128, 1);
    pack_b_k<<<(256 * 256 + 255) / 256, 256, 0, stream>>>(proj_w, pw_p,  256, 256, 0);
    pack_b_k<<<(256 * 1024 + 255) / 256, 256, 0, stream>>>(fc1_w, f1w_p, 256, 1024, 0);
    pack_b_k<<<(512 * 256 + 255) / 256, 256, 0, stream>>>(fc2_w,  f2w_p, 512, 256, 0);

    // ---- SDAM ----
    pool_w_k<<<(BQ * CC * HH + 255) / 256, 256, 0, stream>>>(x, xm3);
    pool_h_k<<<(BQ * CC * WW + 255) / 256, 256, 0, stream>>>(x, xm2);
    dwconv1d_k7_k<<<(BQ * CC * 64 + 255) / 256, 256, 0, stream>>>(xm3, sda_w, c1);
    dwconv1d_k7_k<<<(BQ * CC * 64 + 255) / 256, 256, 0, stream>>>(xm2, sda_w, c2);
    gn_relu_k<<<BQ * 16, 256, 0, stream>>>(c1, sda_gn_w, sda_gn_b, xhb);
    gn_relu_k<<<BQ * 16, 256, 0, stream>>>(c2, sda_gn_w, sda_gn_b, xwb);
    gate_transpose_k<<<(BQ * CC * NN) / 256, 256, 0, stream>>>(x, xhb, xwb, t_res);

    // ---- norm1 ----
    layernorm_k<0, 0><<<MTOK / 4, 128, 0, stream>>>(t_res, CC, 0, n1_w, n1_b, 1e-6f,
                                                    MTOK, CC, t_ln, nullptr);

    // ---- spatial reduction -> K/V ----
    red1_k<<<(BQ * CC * 256 + 255) / 256, 256, 0, stream>>>(t_ln, red_w, red_b, red1b);
    red2_k<<<(BQ * CC * 16 + 255) / 256, 256, 0, stream>>>(red1b, red_w, red_b, red2b);
    dw3_t_k<<<(BQ * CC * 16 + 255) / 256, 256, 0, stream>>>(red2b, dw_w, dw_b, dwt);
    gemm_wmma_k<0, 0><<<dim3((BQ * NRR) / 64, 128 / 64), 128, 0, stream>>>(
        dwt, cw_p, conv_b, nullptr, cx, BQ * NRR, 128, 256, 0);
    layernorm_k<1, 0><<<(BQ * NRR) / 4, 128, 0, stream>>>(cx, CR, 0, na_w, na_b, 1e-5f,
                                                          BQ * NRR, CR, xr, nullptr);
    gemm_wmma_k<0, 0><<<dim3((BQ * NRR) / 64, 128 / 64), 128, 0, stream>>>(
        xr, kw_p, nullptr, nullptr, kbuf, BQ * NRR, 128, 128, 0);
    gemm_wmma_k<0, 0><<<dim3((BQ * NRR) / 64, 256 / 64), 128, 0, stream>>>(
        xr, vw_p, nullptr, nullptr, vbuf, BQ * NRR, 256, 128, 0);
    cpe_k<<<(BQ * CC * 16 + 255) / 256, 256, 0, stream>>>(vbuf, cpe_w, cpe_b, vbuf2);

    // ---- Q projection (big WMMA GEMM) + fused attention ----
    gemm_wmma_k<0, 0><<<dim3(MTOK / 64, 128 / 64), 128, 0, stream>>>(
        t_ln, qw_p, nullptr, nullptr, qbuf, MTOK, 128, 256, 0);
    attn_k<<<BQ * 16, 256, 0, stream>>>(qbuf, kbuf, vbuf2, obuf);

    // ---- output projection + residual ----
    gemm_wmma_k<0, 0><<<dim3(MTOK / 64, 256 / 64), 128, 0, stream>>>(
        obuf, pw_p, proj_b, t_res, t2, MTOK, 256, 256, CC);

    // ---- conv-gated MLP ----
    layernorm_k<0, 0><<<MTOK / 4, 128, 0, stream>>>(t2, CC, 0, n2_w, n2_b, 1e-6f,
                                                    MTOK, CC, t2_ln, nullptr);
    gemm_wmma_k<1, 0><<<dim3(MTOK / 64, 1024 / 64), 128, 0, stream>>>(
        t2_ln, f1w_p, fc1_b, nullptr, hbuf, MTOK, 1024, 256, 0);          // +gelu
    layernorm_k<0, 1><<<MTOK / 4, 128, 0, stream>>>(hbuf, HIDN, HALFN, gn_w, gn_b,
                                                    1e-5f, MTOK, HALFN, nullptr, x2img);
    gconv_mul_k<<<(unsigned)(((size_t)BQ * HALFN * NN) / 256), 256, 0, stream>>>(
        x2img, hbuf, gc_w, gc_b, prod);
    gemm_wmma_k<0, 1><<<dim3(MTOK / 64, 256 / 64), 128, 0, stream>>>(
        prod, f2w_p, fc2_b, t2, out, MTOK, 256, 512, CC);                 // NCHW scatter
}
